// Codebook_26714696581530
// MI455X (gfx1250) — matrix operand order repaired and checked by reading, compiled, not run on hardware
//
#include <hip/hip_runtime.h>

// ---------------- problem constants (match reference) ----------------
#define N_TOK 65536
#define DIM   1024
#define EMB   256
#define KCODE 1024
#define BN_EPS 1e-5f
#define COMMIT 0.25f

typedef _Float16 f16;
typedef __attribute__((ext_vector_type(16))) _Float16 v16h;
typedef __attribute__((ext_vector_type(8)))  _Float16 v8h;
typedef __attribute__((ext_vector_type(8)))  float    v8f;
typedef __attribute__((ext_vector_type(4)))  float    v4f;

static __device__ __forceinline__ v8f wmma_f16(v16h a, v16h b, v8f c) {
  // 8 args: (neg_a, A, neg_b, B, c_mod, C, reuse_a, reuse_b)
  return __builtin_amdgcn_wmma_f32_16x16x32_f16(false, a, false, b, (short)0, c,
                                                false, false);
}

// ---------------- small prep kernels ----------------
__global__ void cvt_f32_to_f16(const float* __restrict__ src, f16* __restrict__ dst, int n) {
  int i = blockIdx.x * blockDim.x + threadIdx.x;
  if (i < n) dst[i] = (f16)src[i];
}

// codebook: emit f16 copy for WMMA B operand, and the fp32 passthrough output
__global__ void emb_prep(const float* __restrict__ emb, f16* __restrict__ embh,
                         float* __restrict__ out_emb) {
  int i = blockIdx.x * blockDim.x + threadIdx.x; // [0, K*E)
  float v = emb[i];
  embh[i] = (f16)v;
  out_emb[i] = v;
}

// per-code squared norm ||e_k||^2
__global__ void e2_kernel(const float* __restrict__ emb, float* __restrict__ e2) {
  int k = blockIdx.x * blockDim.x + threadIdx.x; // [0, K)
  const float* row = emb + (size_t)k * EMB;
  float s = 0.f;
  for (int e = 0; e < EMB; ++e) { float v = row[e]; s += v * v; }
  e2[k] = s;
}

// ---------------- BatchNorm1 statistics over x columns ----------------
#define ROWSPLIT 64
__global__ void bn1_partial(const float* __restrict__ x, float* __restrict__ s,
                            float* __restrict__ s2) {
  int c = blockIdx.x * blockDim.x + threadIdx.x;      // column in [0, DIM)
  int r0 = blockIdx.y * (N_TOK / ROWSPLIT);
  float a = 0.f, b = 0.f;
  for (int r = r0; r < r0 + N_TOK / ROWSPLIT; ++r) {
    float v = x[(size_t)r * DIM + c];
    a += v; b += v * v;
  }
  atomicAdd(&s[c], a);
  atomicAdd(&s2[c], b);
}

__global__ void bn1_final(const float* __restrict__ s, const float* __restrict__ s2,
                          const float* __restrict__ gamma, const float* __restrict__ beta,
                          float* __restrict__ sc, float* __restrict__ sh) {
  int c = blockIdx.x * blockDim.x + threadIdx.x;
  float mean = s[c] * (1.f / N_TOK);
  float var  = s2[c] * (1.f / N_TOK) - mean * mean;
  float rs   = rsqrtf(var + BN_EPS);
  float scale = gamma[c] * rs;
  sc[c] = scale;
  sh[c] = beta[c] - mean * scale;
}

// A-fragment builder: 16 f32 (BN-normalized) -> f16, contiguous k order: element s holds k = kb + s (B loads match).
static __device__ __forceinline__ v16h build_a_frag(const float* __restrict__ row,
                                                    const float* __restrict__ sc,
                                                    const float* __restrict__ sh,
                                                    int kb) {
  v4f x0 = *(const v4f*)(row + kb);
  v4f x1 = *(const v4f*)(row + kb + 4);
  v4f x2 = *(const v4f*)(row + kb + 8);
  v4f x3 = *(const v4f*)(row + kb + 12);
  v4f s0 = *(const v4f*)(sc + kb),      h0 = *(const v4f*)(sh + kb);
  v4f s1 = *(const v4f*)(sc + kb + 4),  h1 = *(const v4f*)(sh + kb + 4);
  v4f s2 = *(const v4f*)(sc + kb + 8),  h2 = *(const v4f*)(sh + kb + 8);
  v4f s3 = *(const v4f*)(sc + kb + 12), h3 = *(const v4f*)(sh + kb + 12);
  v16h a;
#pragma unroll
  for (int i = 0; i < 4; ++i) {
    a[i]      = (f16)(x0[i] * s0[i] + h0[i]);
    a[4 + i]  = (f16)(x1[i] * s1[i] + h1[i]);
    a[8 + i]  = (f16)(x2[i] * s2[i] + h2[i]);
    a[12 + i] = (f16)(x3[i] * s3[i] + h3[i]);
  }
  return a;
}

// ---------------- GEMM1: h16 = f16( BN(x) @ w1^T + b1 ) ----------------
// 8 waves/block: 4 row-groups x 2 col-groups. Wave: 16 rows x 128 cols
// (8 col tiles -> 64 acc VGPRs, no spills). Col-wave pairs share x rows (L2/WGP$).
__global__ void gemm1_kernel(const float* __restrict__ x, const f16* __restrict__ w1h,
                             const float* __restrict__ b1, const float* __restrict__ sc1,
                             const float* __restrict__ sh1, f16* __restrict__ h16) {
  int wave = threadIdx.x >> 5, lane = threadIdx.x & 31;
  int rg = wave >> 1, cg = wave & 1;
  int m0 = blockIdx.x * 64 + rg * 16;
  int half = lane >> 4, ml = lane & 15;
  int nbase = cg * 128 + ml;                 // n = nbase + t*16, t in [0,8)

  v8f acc[8];
#pragma unroll
  for (int t = 0; t < 8; ++t) acc[t] = (v8f)0.f;

  const float* xrow = x + (size_t)(m0 + ml) * DIM;

  for (int k0 = 0; k0 < DIM; k0 += 32) {
    v16h a = build_a_frag(xrow, sc1, sh1, k0 + half * 16);
    const f16* bptr = w1h + (size_t)nbase * DIM + k0 + half * 16;
    v16h b = *(const v16h*)bptr;             // double-buffered B fragments
#pragma unroll
    for (int t = 0; t < 8; ++t) {
      v16h bn;
      if (t < 7) bn = *(const v16h*)(bptr + (size_t)(t + 1) * 16 * DIM);
      acc[t] = wmma_f16(a, b, acc[t]);
      if (t < 7) b = bn;
    }
  }
  // epilogue: + b1, convert to f16
#pragma unroll
  for (int t = 0; t < 8; ++t) {
    int n = nbase + t * 16;
    float bias = b1[n];
#pragma unroll
    for (int r = 0; r < 8; ++r) {
      int m = m0 + half * 8 + r;
      h16[(size_t)m * EMB + n] = (f16)(acc[t][r] + bias);
    }
  }
}

// ---------------- distance + argmin (streamed, no NxK matrix) ----------------
__global__ void dist_argmin_kernel(const f16* __restrict__ h16, const f16* __restrict__ embh,
                                   const float* __restrict__ e2, int* __restrict__ idx,
                                   int* __restrict__ counts) {
  int wave = threadIdx.x >> 5, lane = threadIdx.x & 31;
  int m0 = blockIdx.x * 128 + wave * 16;
  int half = lane >> 4, ml = lane & 15;

  // preload all 8 A-chunks of this wave's 16 h-rows (EMB=256 -> 8 x K32)
  v16h A[8];
  const f16* hrow = h16 + (size_t)(m0 + ml) * EMB;
#pragma unroll
  for (int c = 0; c < 8; ++c) {
    int kb = c * 32 + half * 16;
    v8h lo = *(const v8h*)(hrow + kb);
    v8h hi = *(const v8h*)(hrow + kb + 8);
    v16h a;
#pragma unroll
    for (int i = 0; i < 8; ++i) { a[i] = lo[i]; a[8 + i] = hi[i]; }
    A[c] = a;
  }

  float bs[8]; int bi[8];
#pragma unroll
  for (int r = 0; r < 8; ++r) { bs[r] = 3.4e38f; bi[r] = 0; }

  for (int kt = 0; kt < KCODE / 16; ++kt) {
    int n = kt * 16 + ml;                       // candidate code id handled by lane
    const f16* brow = embh + (size_t)n * EMB + half * 16;
    v8f acc = (v8f)0.f;
    v16h b = *(const v16h*)brow;               // double-buffered B fragments
#pragma unroll
    for (int c = 0; c < 8; ++c) {
      v16h bn;
      if (c < 7) bn = *(const v16h*)(brow + (c + 1) * 32);
      acc = wmma_f16(A[c], b, acc);
      if (c < 7) b = bn;
    }
    float en = e2[n];
#pragma unroll
    for (int r = 0; r < 8; ++r) {
      float s = en - 2.f * acc[r];             // ||h||^2 is row-constant: dropped
      if (s < bs[r]) { bs[r] = s; bi[r] = n; }
      else if (s == bs[r] && n < bi[r]) { bi[r] = n; }
    }
  }
  // reduce across the 16 lanes of each half-wave (xor<16 stays in-group)
#pragma unroll
  for (int r = 0; r < 8; ++r) {
#pragma unroll
    for (int off = 8; off >= 1; off >>= 1) {
      float os = __shfl_xor(bs[r], off, 32);
      int   oi = __shfl_xor(bi[r], off, 32);
      if (os < bs[r] || (os == bs[r] && oi < bi[r])) { bs[r] = os; bi[r] = oi; }
    }
  }
  if (ml == 0) {                                 // lanes 0 and 16
#pragma unroll
    for (int r = 0; r < 8; ++r) {
      int m = m0 + half * 8 + r;
      idx[m] = bi[r];
      atomicAdd(&counts[bi[r]], 1);
    }
  }
}

// ---------------- BN2 stats analytically from usage histogram ----------------
__global__ void bn2_kernel(const int* __restrict__ counts, const float* __restrict__ emb,
                           const float* __restrict__ gamma, const float* __restrict__ beta,
                           float* __restrict__ sc, float* __restrict__ sh) {
  int e = threadIdx.x;                            // 256 threads, 1 block
  float s = 0.f, s2 = 0.f;
  for (int k = 0; k < KCODE; ++k) {
    float cnt = (float)counts[k];
    float w = emb[(size_t)k * EMB + e];
    s += cnt * w;
    s2 += cnt * w * w;
  }
  float mean = s * (1.f / N_TOK);
  float var  = s2 * (1.f / N_TOK) - mean * mean;
  float rs   = rsqrtf(var + BN_EPS);
  float scale = gamma[e] * rs;
  sc[e] = scale;
  sh[e] = beta[e] - mean * scale;
}

// ---------------- GEMM2: quantized = BN(gather) @ w2^T + b2 ; SSE vs x -------
__global__ void gemm2_kernel(const float* __restrict__ emb, const int* __restrict__ idx,
                             const f16* __restrict__ w2h, const float* __restrict__ b2,
                             const float* __restrict__ sc2, const float* __restrict__ sh2,
                             const float* __restrict__ x, float* __restrict__ outq,
                             float* __restrict__ sse) {
  int wave = threadIdx.x >> 5, lane = threadIdx.x & 31;
  int m0 = blockIdx.x * 128 + wave * 16;
  int half = lane >> 4, ml = lane & 15;
  int row = m0 + ml;
  const float* erow = emb + (size_t)idx[row] * EMB;  // gather the selected code

  // preload A: normalized gathered code rows, 8 chunks of K=32
  v16h A[8];
#pragma unroll
  for (int c = 0; c < 8; ++c) {
    A[c] = build_a_frag(erow, sc2, sh2, c * 32 + half * 16);
  }

  float lsse = 0.f;
  for (int dt = 0; dt < DIM / 16; ++dt) {
    int n = dt * 16 + ml;
    const f16* brow = w2h + (size_t)n * EMB + half * 16;
    v8f acc = (v8f)0.f;
    v16h b = *(const v16h*)brow;               // double-buffered B fragments
#pragma unroll
    for (int c = 0; c < 8; ++c) {
      v16h bn;
      if (c < 7) bn = *(const v16h*)(brow + (c + 1) * 32);
      acc = wmma_f16(A[c], b, acc);
      if (c < 7) b = bn;
    }
    float bias = b2[n];
#pragma unroll
    for (int r = 0; r < 8; ++r) {
      int m = m0 + half * 8 + r;
      float v = acc[r] + bias;
      outq[(size_t)m * DIM + n] = v;            // quantized == q numerically
      float d = v - x[(size_t)m * DIM + n];
      lsse += d * d;
    }
  }
#pragma unroll
  for (int off = 16; off >= 1; off >>= 1) lsse += __shfl_xor(lsse, off, 32);
  if (lane == 0) atomicAdd(sse, lsse);
}

// ---------------- usage + loss finalize ----------------
__global__ void finalize_kernel(const int* __restrict__ counts, const float* __restrict__ sse,
                                float* __restrict__ usage, float* __restrict__ loss) {
  int k = blockIdx.x * blockDim.x + threadIdx.x;
  if (k < KCODE) usage[k] = (float)counts[k] * (1.f / N_TOK);
  if (k == 0) {
    float mse = sse[0] * (1.f / ((float)N_TOK * (float)DIM));
    loss[0] = (1.f + COMMIT) * mse;             // q_latent + COMMIT * e_latent
  }
}

// ---------------- host orchestration ----------------
extern "C" void kernel_launch(void* const* d_in, const int* in_sizes, int n_in,
                              void* d_out, int out_size, void* d_ws, size_t ws_size,
                              hipStream_t stream) {
  const float* x      = (const float*)d_in[0];
  const float* emb_w  = (const float*)d_in[1];
  const float* w1     = (const float*)d_in[2];
  const float* b1     = (const float*)d_in[3];
  const float* w2     = (const float*)d_in[4];
  const float* b2     = (const float*)d_in[5];
  const float* gamma1 = (const float*)d_in[6];
  const float* beta1  = (const float*)d_in[7];
  const float* gamma2 = (const float*)d_in[8];
  const float* beta2  = (const float*)d_in[9];

  float* out      = (float*)d_out;
  float* out_loss = out;
  float* out_q    = out + 1;
  float* out_use  = out + 1 + (size_t)N_TOK * DIM;
  float* out_emb  = out_use + KCODE;

  // workspace carve (256B aligned)
  char* base = (char*)d_ws;
  size_t cur = 0;
  auto carve = [&](size_t bytes) -> char* {
    char* p = base + cur;
    cur = (cur + bytes + 255) & ~(size_t)255;
    return p;
  };
  f16*   w1h    = (f16*)carve((size_t)EMB * DIM * 2);
  f16*   w2h    = (f16*)carve((size_t)DIM * EMB * 2);
  f16*   embh   = (f16*)carve((size_t)KCODE * EMB * 2);
  f16*   h16    = (f16*)carve((size_t)N_TOK * EMB * 2);
  int*   idx    = (int*)carve((size_t)N_TOK * 4);
  int*   counts = (int*)carve((size_t)KCODE * 4);
  float* s1     = (float*)carve((size_t)DIM * 4);
  float* s1sq   = (float*)carve((size_t)DIM * 4);
  float* sc1    = (float*)carve((size_t)DIM * 4);
  float* sh1    = (float*)carve((size_t)DIM * 4);
  float* sc2    = (float*)carve((size_t)EMB * 4);
  float* sh2    = (float*)carve((size_t)EMB * 4);
  float* e2     = (float*)carve((size_t)KCODE * 4);
  float* sse    = (float*)carve(256);
  (void)ws_size; (void)in_sizes; (void)n_in; (void)out_size;

  // zero accumulators (capturable stream memsets)
  hipMemsetAsync(s1, 0, (size_t)DIM * 4, stream);
  hipMemsetAsync(s1sq, 0, (size_t)DIM * 4, stream);
  hipMemsetAsync(counts, 0, (size_t)KCODE * 4, stream);
  hipMemsetAsync(sse, 0, sizeof(float), stream);

  // weight / codebook prep
  cvt_f32_to_f16<<<(EMB * DIM) / 256, 256, 0, stream>>>(w1, w1h, EMB * DIM);
  cvt_f32_to_f16<<<(DIM * EMB) / 256, 256, 0, stream>>>(w2, w2h, DIM * EMB);
  emb_prep<<<(KCODE * EMB) / 256, 256, 0, stream>>>(emb_w, embh, out_emb);
  e2_kernel<<<KCODE / 256, 256, 0, stream>>>(emb_w, e2);

  // BN1 stats
  bn1_partial<<<dim3(DIM / 256, ROWSPLIT), 256, 0, stream>>>(x, s1, s1sq);
  bn1_final<<<DIM / 256, 256, 0, stream>>>(s1, s1sq, gamma1, beta1, sc1, sh1);

  // GEMM1: h = BN(x) @ w1^T + b1  (8 waves/block: 4 row-groups x 2 col-groups)
  gemm1_kernel<<<N_TOK / 64, 256, 0, stream>>>(x, w1h, b1, sc1, sh1, h16);

  // nearest-code search (WMMA distances + register argmin + histogram)
  dist_argmin_kernel<<<N_TOK / 128, 256, 0, stream>>>(h16, embh, e2, idx, counts);

  // BN2 stats from histogram
  bn2_kernel<<<1, EMB, 0, stream>>>(counts, emb_w, gamma2, beta2, sc2, sh2);

  // GEMM2: quantized = BN(gather) @ w2^T + b2 ; accumulate SSE vs x
  gemm2_kernel<<<N_TOK / 128, 256, 0, stream>>>(emb_w, idx, w2h, b2, sc2, sh2,
                                                x, out_q, sse);

  // usage + loss
  finalize_kernel<<<KCODE / 256, 256, 0, stream>>>(counts, sse, out_use, out_loss);
}
